// MultiHeadAttention_62380105007539
// MI455X (gfx1250) — matrix operand order repaired and checked by reading, not compiled, not run on hardware
//
#include <hip/hip_runtime.h>

#define DMODEL 1024
#define HEADS  16
#define KSZ    64
#define SEQ    2048
#define BATCH  2
#define MROWS  (BATCH * SEQ)     // 4096

typedef __attribute__((ext_vector_type(16))) __bf16        v16bf;
typedef __attribute__((ext_vector_type(8)))  float         v8f;
typedef __attribute__((ext_vector_type(4)))  unsigned int  v4u;
typedef __attribute__((ext_vector_type(4)))  unsigned int  u32x4;
typedef __attribute__((ext_vector_type(8)))  int           i32x8;
typedef __attribute__((ext_vector_type(4)))  int           i32x4;

union Frag16 { v16bf v; v4u q[2]; };

__device__ inline unsigned short f2bf(float f) {
  return __builtin_bit_cast(unsigned short, (__bf16)f);
}

__device__ inline v8f wmma_bf16(const Frag16& a, const Frag16& b, v8f c) {
  return __builtin_amdgcn_wmma_f32_16x16x32_bf16(false, a.v, false, b.v,
                                                 (short)0, c, false, false);
}

// ---------------------------------------------------------------------------
// TDM: async DMA of a 2-D bf16 tile (tile_w x tile_h) from a row-major tensor
// (row stride in elements) into LDS at lds_off, row-major with row stride
// tile_w.  Descriptor per CDNA5 ISA 8.3/8.4.  Issue from one wave; tracked by
// TENSORcnt.  This toolchain uses the 6-arg builtin form.
// ---------------------------------------------------------------------------
__device__ inline void tdm_load_2d(unsigned lds_off, const void* gaddr,
                                   unsigned tensor_w, unsigned tensor_h,
                                   unsigned long long row_stride,
                                   unsigned tile_w, unsigned tile_h)
{
  unsigned long long ga = (unsigned long long)(size_t)gaddr;
  u32x4 g0;
  g0[0] = 1u;                                        // count=1 (user D#)
  g0[1] = lds_off;                                   // lds_addr (bytes)
  g0[2] = (unsigned)(ga & 0xFFFFFFFFu);              // global_addr[31:0]
  g0[3] = (unsigned)((ga >> 32) & 0x1FFFFFFu)        // global_addr[56:32]
        | 0x80000000u;                               // type=2 ("image")
  i32x8 g1;
  g1[0] = (int)(1u << 16);                           // data_size=1 -> 2 bytes
  g1[1] = (int)((tensor_w & 0xFFFFu) << 16);         // tensor_dim0[15:0]
  g1[2] = (int)((tensor_w >> 16) | ((tensor_h & 0xFFFFu) << 16));
  g1[3] = (int)((tensor_h >> 16) | (tile_w << 16));  // dim1 hi | tile_dim0
  g1[4] = (int)(tile_h & 0xFFFFu);                   // tile_dim1 (tile_dim2=0)
  g1[5] = (int)(unsigned)(row_stride & 0xFFFFFFFFull);      // dim0_stride lo
  g1[6] = (int)(unsigned)((row_stride >> 32) & 0xFFFFull);  // dim0_stride hi
  g1[7] = 0;
  i32x4 z4 = {0, 0, 0, 0};
  i32x8 z8 = {0, 0, 0, 0, 0, 0, 0, 0};
  __builtin_amdgcn_tensor_load_to_lds(g0, g1, z4, z4, z8, 0);
}

__device__ inline unsigned lds_off_of(const void* p) {
  return (unsigned)(size_t)p;        // flat LDS address: low 32 bits = offset
}

// ---------------------------------------------------------------------------
// Kernel 0: one-time convert/transpose pass.
//   z=0 : xbf  = bf16(x)                       (4096 x 1024, natural)
//   z=1..4 : Wt = bf16(W^T)  [n][k]            (1024 x 1024)
// ---------------------------------------------------------------------------
__global__ __launch_bounds__(256) void convert_pack(
    const float* __restrict__ x,
    const float* __restrict__ Wq, const float* __restrict__ Wk,
    const float* __restrict__ Wv, const float* __restrict__ Wo,
    unsigned short* __restrict__ xbf,
    unsigned short* __restrict__ wqt, unsigned short* __restrict__ wkt,
    unsigned short* __restrict__ wvt, unsigned short* __restrict__ wot)
{
  const int z = blockIdx.z;
  if (z == 0) {
    size_t base = (size_t)blockIdx.x * 4096;
    for (int i = 0; i < 16; ++i) {
      size_t idx = base + (size_t)i * 256 + threadIdx.x;
      xbf[idx] = f2bf(x[idx]);
    }
  } else {
    if (blockIdx.x >= 256) return;
    const float* W = (z == 1) ? Wq : (z == 2) ? Wk : (z == 3) ? Wv : Wo;
    unsigned short* T = (z == 1) ? wqt : (z == 2) ? wkt : (z == 3) ? wvt : wot;
    size_t base = (size_t)blockIdx.x * 4096;
    for (int i = 0; i < 16; ++i) {
      size_t idx = base + (size_t)i * 256 + threadIdx.x;  // n-major out index
      int n = (int)(idx >> 10), k = (int)(idx & 1023);
      T[idx] = f2bf(W[(size_t)k * DMODEL + n]);
    }
  }
}

// ---------------------------------------------------------------------------
// Kernel 1: QKV projection, TDM-staged, double-buffered.
// Block = 128 thr (4 waves) computes a 128x64 tile; each wave 32 rows.
// Writes Q (x0.125) and K in (B,H,S,64); V transposed in (B,H,64,S).
// ---------------------------------------------------------------------------
__global__ __launch_bounds__(128) void qkv_gemm(
    const unsigned short* __restrict__ xbf,
    const unsigned short* __restrict__ wqt, const unsigned short* __restrict__ wkt,
    const unsigned short* __restrict__ wvt,
    const float* __restrict__ bq, const float* __restrict__ bk,
    const float* __restrict__ bv,
    unsigned short* __restrict__ qws, unsigned short* __restrict__ kws,
    unsigned short* __restrict__ vws)
{
  __shared__ __align__(16) unsigned short Asb[2][128 * 32]; // [m][k]
  __shared__ __align__(16) unsigned short Bsb[2][64 * 32];  // [n][k]

  const int tid  = threadIdx.x;
  const int lane = tid & 31, wave = tid >> 5;
  const int col16 = lane & 15, hi = lane >> 4;
  const int n0 = blockIdx.x * 64, m0 = blockIdx.y * 128;
  const int zz = blockIdx.z;
  const unsigned short* Wt = (zz == 0) ? wqt : (zz == 1) ? wkt : wvt;
  const float* bias = (zz == 0) ? bq : (zz == 1) ? bk : bv;
  unsigned short* out = (zz == 0) ? qws : (zz == 1) ? kws : vws;
  const float oscale = (zz == 0) ? 0.125f : 1.0f;

  v8f acc[2][4];
#pragma unroll
  for (int mi = 0; mi < 2; ++mi)
#pragma unroll
    for (int f = 0; f < 4; ++f)
#pragma unroll
      for (int r = 0; r < 8; ++r) acc[mi][f][r] = 0.0f;

  if (wave == 0) {
    tdm_load_2d(lds_off_of(&Asb[0][0]), xbf + (size_t)m0 * DMODEL,
                DMODEL, MROWS, DMODEL, 32, 128);
    tdm_load_2d(lds_off_of(&Bsb[0][0]), Wt + (size_t)n0 * DMODEL,
                DMODEL, DMODEL, DMODEL, 32, 64);
  }

  for (int kt = 0; kt < DMODEL; kt += 32) {
    const int buf = (kt >> 5) & 1;
    if (wave == 0) {
      if (kt + 32 < DMODEL) {
        tdm_load_2d(lds_off_of(&Asb[buf ^ 1][0]), xbf + (size_t)m0 * DMODEL + kt + 32,
                    DMODEL, MROWS, DMODEL, 32, 128);
        tdm_load_2d(lds_off_of(&Bsb[buf ^ 1][0]), Wt + (size_t)n0 * DMODEL + kt + 32,
                    DMODEL, DMODEL, DMODEL, 32, 64);
        __builtin_amdgcn_s_wait_tensorcnt(2);
      } else {
        __builtin_amdgcn_s_wait_tensorcnt(0);
      }
    }
    __syncthreads();

    Frag16 a[2], b[4];
#pragma unroll
    for (int mi = 0; mi < 2; ++mi) {
      const int rowb = (wave * 32 + mi * 16 + col16) * 32;
      a[mi].q[0] = *(const v4u*)&Asb[buf][rowb + hi * 8];
      a[mi].q[1] = *(const v4u*)&Asb[buf][rowb + 16 + hi * 8];
    }
#pragma unroll
    for (int f = 0; f < 4; ++f) {
      const int rowb = (f * 16 + col16) * 32;
      b[f].q[0] = *(const v4u*)&Bsb[buf][rowb + hi * 8];
      b[f].q[1] = *(const v4u*)&Bsb[buf][rowb + 16 + hi * 8];
    }
#pragma unroll
    for (int mi = 0; mi < 2; ++mi)
#pragma unroll
      for (int f = 0; f < 4; ++f)
        acc[mi][f] = wmma_bf16(a[mi], b[f], acc[mi][f]);
    __syncthreads();
  }

#pragma unroll
  for (int f = 0; f < 4; ++f) {
    const int col = n0 + f * 16 + col16;
    const float bv_ = bias[col];
    const int h = col >> 6, kk = col & 63;
#pragma unroll
    for (int mi = 0; mi < 2; ++mi)
#pragma unroll
      for (int r = 0; r < 8; ++r) {
        const int row = m0 + wave * 32 + mi * 16 + r + 8 * hi;
        const int bb = row >> 11, s = row & 2047;
        const size_t dst = (zz == 2)
            ? ((size_t)(bb * HEADS + h) * KSZ + kk) * SEQ + s      // V^T
            : ((size_t)(bb * HEADS + h) * SEQ + s) * KSZ + kk;     // Q, K
        out[dst] = f2bf((acc[mi][f][r] + bv_) * oscale);
      }
  }
}

// ---------------------------------------------------------------------------
// Kernel 2: flash attention.  K tile and V^T tile are TDM double-buffered;
// per 64-col block: 8 WMMA scores, online softmax, P through per-wave LDS,
// 8 WMMA for P@V.
// ---------------------------------------------------------------------------
__global__ __launch_bounds__(128) void flash_attn(
    const unsigned short* __restrict__ qws,
    const unsigned short* __restrict__ kws,
    const unsigned short* __restrict__ vws,   // (B,H,64,S) transposed
    unsigned short* __restrict__ ows)
{
  __shared__ __align__(16) unsigned short Ks[2][64 * 64];   // [t][kk]
  __shared__ __align__(16) unsigned short Vts[2][64 * 64];  // [kk][t]
  __shared__ __align__(16) unsigned short Ps[4][16 * 64];

  const int tid  = threadIdx.x;
  const int lane = tid & 31, wave = tid >> 5;
  const int col16 = lane & 15, hi = lane >> 4;
  const int bh = blockIdx.y;
  const int s0 = blockIdx.x * 64;
  const size_t headbase = (size_t)bh * SEQ * KSZ;
  const unsigned short* kptr = kws + headbase;
  const unsigned short* vtptr = vws + headbase;   // [64][2048]

  Frag16 aq[2];
  {
    const unsigned short* qrow =
        qws + headbase + (size_t)(s0 + wave * 16 + col16) * KSZ;
#pragma unroll
    for (int c = 0; c < 2; ++c) {
      aq[c].q[0] = *(const v4u*)&qrow[c * 32 + hi * 8];
      aq[c].q[1] = *(const v4u*)&qrow[c * 32 + 16 + hi * 8];
    }
  }

  float mrow[8], lrow[8];
  v8f oacc[4];
#pragma unroll
  for (int r = 0; r < 8; ++r) { mrow[r] = -3.0e38f; lrow[r] = 0.0f; }
#pragma unroll
  for (int f = 0; f < 4; ++f)
#pragma unroll
    for (int r = 0; r < 8; ++r) oacc[f][r] = 0.0f;

  if (wave == 0) {
    tdm_load_2d(lds_off_of(&Ks[0][0]),  kptr,  KSZ, SEQ, KSZ, 64, 64);
    tdm_load_2d(lds_off_of(&Vts[0][0]), vtptr, SEQ, KSZ, SEQ, 64, 64);
  }

  for (int t0 = 0; t0 < SEQ; t0 += 64) {
    const int buf = (t0 >> 6) & 1;
    if (wave == 0) {
      if (t0 + 64 < SEQ) {
        tdm_load_2d(lds_off_of(&Ks[buf ^ 1][0]),
                    kptr + (size_t)(t0 + 64) * KSZ, KSZ, SEQ, KSZ, 64, 64);
        tdm_load_2d(lds_off_of(&Vts[buf ^ 1][0]),
                    vtptr + (t0 + 64), SEQ, KSZ, SEQ, 64, 64);
        __builtin_amdgcn_s_wait_tensorcnt(2);
      } else {
        __builtin_amdgcn_s_wait_tensorcnt(0);
      }
    }
    __syncthreads();

    // scores S(16x64) = Q @ K^T
    v8f sc[4];
#pragma unroll
    for (int f = 0; f < 4; ++f)
#pragma unroll
      for (int r = 0; r < 8; ++r) sc[f][r] = 0.0f;
#pragma unroll
    for (int c = 0; c < 2; ++c) {
      Frag16 b[4];
#pragma unroll
      for (int f = 0; f < 4; ++f) {
        const int rowb = (f * 16 + col16) * 64 + c * 32;
        b[f].q[0] = *(const v4u*)&Ks[buf][rowb + hi * 8];
        b[f].q[1] = *(const v4u*)&Ks[buf][rowb + 16 + hi * 8];
      }
#pragma unroll
      for (int f = 0; f < 4; ++f) sc[f] = wmma_bf16(aq[c], b[f], sc[f]);
    }

    // online softmax (row r lives in the 16 lanes of this half-wave)
    float mnew[8], scale[8], rsum[8];
#pragma unroll
    for (int r = 0; r < 8; ++r) {
      float v = fmaxf(fmaxf(sc[0][r], sc[1][r]), fmaxf(sc[2][r], sc[3][r]));
      for (int m = 1; m < 16; m <<= 1) v = fmaxf(v, __shfl_xor(v, m, 32));
      mnew[r]  = fmaxf(mrow[r], v);
      scale[r] = __expf(mrow[r] - mnew[r]);
      rsum[r]  = 0.0f;
    }
#pragma unroll
    for (int f = 0; f < 4; ++f)
#pragma unroll
      for (int r = 0; r < 8; ++r) {
        float p = __expf(sc[f][r] - mnew[r]);
        rsum[r] += p;
        Ps[wave][(r + 8 * hi) * 64 + f * 16 + col16] = f2bf(p);
      }
#pragma unroll
    for (int r = 0; r < 8; ++r) {
      float v = rsum[r];
      for (int m = 1; m < 16; m <<= 1) v += __shfl_xor(v, m, 32);
      lrow[r] = lrow[r] * scale[r] + v;
      mrow[r] = mnew[r];
#pragma unroll
      for (int f = 0; f < 4; ++f) oacc[f][r] *= scale[r];
    }

    // O += P @ V
#pragma unroll
    for (int c = 0; c < 2; ++c) {
      Frag16 ap, b[4];
      ap.q[0] = *(const v4u*)&Ps[wave][col16 * 64 + c * 32 + hi * 8];
      ap.q[1] = *(const v4u*)&Ps[wave][col16 * 64 + c * 32 + 16 + hi * 8];
#pragma unroll
      for (int f = 0; f < 4; ++f) {
        const int rowb = (f * 16 + col16) * 64 + c * 32;
        b[f].q[0] = *(const v4u*)&Vts[buf][rowb + hi * 8];
        b[f].q[1] = *(const v4u*)&Vts[buf][rowb + 16 + hi * 8];
      }
#pragma unroll
      for (int f = 0; f < 4; ++f) oacc[f] = wmma_bf16(ap, b[f], oacc[f]);
    }
    __syncthreads();
  }

#pragma unroll
  for (int r = 0; r < 8; ++r) {
    const float inv = 1.0f / lrow[r];
    const int row = s0 + wave * 16 + r + 8 * hi;
#pragma unroll
    for (int f = 0; f < 4; ++f)
      ows[headbase + (size_t)row * KSZ + f * 16 + col16] =
          f2bf(oacc[f][r] * inv);
  }
}

// ---------------------------------------------------------------------------
// Kernel 3: output projection y = O @ Wo + bo  (fp32 out), TDM-staged.
// ---------------------------------------------------------------------------
__global__ __launch_bounds__(128) void out_gemm(
    const unsigned short* __restrict__ ows,
    const unsigned short* __restrict__ wot,
    const float* __restrict__ bo, float* __restrict__ y)
{
  __shared__ __align__(16) unsigned short Asb[2][128 * 32];
  __shared__ __align__(16) unsigned short Bsb[2][64 * 32];

  const int tid  = threadIdx.x;
  const int lane = tid & 31, wave = tid >> 5;
  const int col16 = lane & 15, hi = lane >> 4;
  const int n0 = blockIdx.x * 64, m0 = blockIdx.y * 128;
  const int bb = m0 >> 11, s0l = m0 & 2047;

  v8f acc[2][4];
#pragma unroll
  for (int mi = 0; mi < 2; ++mi)
#pragma unroll
    for (int f = 0; f < 4; ++f)
#pragma unroll
      for (int r = 0; r < 8; ++r) acc[mi][f][r] = 0.0f;

  // A tile source for k-chunk kt: head h = kt>>6, col kkb = kt&63
  auto asrc = [&](int kt) -> const unsigned short* {
    const int h = kt >> 6, kkb = kt & 63;
    return ows + ((size_t)(bb * HEADS + h) * SEQ + s0l) * KSZ + kkb;
  };

  if (wave == 0) {
    tdm_load_2d(lds_off_of(&Asb[0][0]), asrc(0), KSZ, SEQ, KSZ, 32, 128);
    tdm_load_2d(lds_off_of(&Bsb[0][0]), wot + (size_t)n0 * DMODEL,
                DMODEL, DMODEL, DMODEL, 32, 64);
  }

  for (int kt = 0; kt < DMODEL; kt += 32) {
    const int buf = (kt >> 5) & 1;
    if (wave == 0) {
      if (kt + 32 < DMODEL) {
        tdm_load_2d(lds_off_of(&Asb[buf ^ 1][0]), asrc(kt + 32),
                    KSZ, SEQ, KSZ, 32, 128);
        tdm_load_2d(lds_off_of(&Bsb[buf ^ 1][0]),
                    wot + (size_t)n0 * DMODEL + kt + 32,
                    DMODEL, DMODEL, DMODEL, 32, 64);
        __builtin_amdgcn_s_wait_tensorcnt(2);
      } else {
        __builtin_amdgcn_s_wait_tensorcnt(0);
      }
    }
    __syncthreads();

    Frag16 a[2], b[4];
#pragma unroll
    for (int mi = 0; mi < 2; ++mi) {
      const int rowb = (wave * 32 + mi * 16 + col16) * 32;
      a[mi].q[0] = *(const v4u*)&Asb[buf][rowb + hi * 8];
      a[mi].q[1] = *(const v4u*)&Asb[buf][rowb + 16 + hi * 8];
    }
#pragma unroll
    for (int f = 0; f < 4; ++f) {
      const int rowb = (f * 16 + col16) * 32;
      b[f].q[0] = *(const v4u*)&Bsb[buf][rowb + hi * 8];
      b[f].q[1] = *(const v4u*)&Bsb[buf][rowb + 16 + hi * 8];
    }
#pragma unroll
    for (int mi = 0; mi < 2; ++mi)
#pragma unroll
      for (int f = 0; f < 4; ++f)
        acc[mi][f] = wmma_bf16(a[mi], b[f], acc[mi][f]);
    __syncthreads();
  }

#pragma unroll
  for (int f = 0; f < 4; ++f) {
    const int col = n0 + f * 16 + col16;
    const float bias = bo[col];
#pragma unroll
    for (int mi = 0; mi < 2; ++mi)
#pragma unroll
      for (int r = 0; r < 8; ++r) {
        const int row = m0 + wave * 32 + mi * 16 + r + 8 * hi;
        y[(size_t)row * DMODEL + col] = acc[mi][f][r] + bias;
      }
  }
}

// ---------------------------------------------------------------------------
extern "C" void kernel_launch(void* const* d_in, const int* in_sizes, int n_in,
                              void* d_out, int out_size, void* d_ws, size_t ws_size,
                              hipStream_t stream) {
  const float* x  = (const float*)d_in[0];
  const float* Wq = (const float*)d_in[1];
  const float* bq = (const float*)d_in[2];
  const float* Wk = (const float*)d_in[3];
  const float* bk = (const float*)d_in[4];
  const float* Wv = (const float*)d_in[5];
  const float* bv = (const float*)d_in[6];
  const float* Wo = (const float*)d_in[7];
  const float* bo = (const float*)d_in[8];
  float* y = (float*)d_out;

  const size_t per = (size_t)BATCH * HEADS * SEQ * KSZ;   // 4,194,304
  const size_t wsz = (size_t)DMODEL * DMODEL;             // 1,048,576
  unsigned short* qws = (unsigned short*)d_ws;
  unsigned short* kws = qws + per;
  unsigned short* vws = kws + per;        // transposed (B,H,64,S)
  unsigned short* ows = vws + per;
  unsigned short* xbf = ows + per;
  unsigned short* wqt = xbf + per;
  unsigned short* wkt = wqt + wsz;
  unsigned short* wvt = wkt + wsz;
  unsigned short* wot = wvt + wsz;

  convert_pack<<<dim3(1024, 1, 5), 256, 0, stream>>>(x, Wq, Wk, Wv, Wo,
                                                     xbf, wqt, wkt, wvt, wot);
  qkv_gemm<<<dim3(16, 32, 3), 128, 0, stream>>>(xbf, wqt, wkt, wvt,
                                                bq, bk, bv, qws, kws, vws);
  flash_attn<<<dim3(SEQ / 64, BATCH * HEADS), 128, 0, stream>>>(qws, kws, vws,
                                                                ows);
  out_gemm<<<dim3(16, 32), 128, 0, stream>>>(ows, wot, bo, y);
}
